// MHAttention_74371653697592
// MI455X (gfx1250) — compile-verified, matrix operand order checked by reading, not run on hardware
//
#include <hip/hip_runtime.h>

typedef __attribute__((ext_vector_type(16))) __bf16 v16bf;
typedef __attribute__((ext_vector_type(8)))  __bf16 v8bf;
typedef __attribute__((ext_vector_type(4)))  __bf16 v4bf;
typedef __attribute__((ext_vector_type(8)))  float  v8f;
typedef __attribute__((ext_vector_type(4)))  float  v4f;

#define TILE_M 64
#define TILE_N 64
#define TILE_K 32

// ---- CDNA5 async global->LDS 16B copy (VGLOBAL async: VDST = LDS byte-address
//      VGPR, VADDR = 64-bit global address, SADDR = off). LDS operand = low 32
//      bits of the flat shared pointer (ISA §10.2 aperture mapping).
#define ASYNC_CP16(g, l)                                                     \
    asm volatile("global_load_async_to_lds_b128 %0, %1, off"                 \
                 :: "v"((unsigned)(unsigned long long)(l)),                  \
                    "v"((const void*)(g))                                    \
                 : "memory")
#if __has_builtin(__builtin_amdgcn_s_wait_asynccnt)
#define ASYNC_WAIT(n) __builtin_amdgcn_s_wait_asynccnt(n)
#else
#define ASYNC_WAIT(n) asm volatile("s_wait_asynccnt %0" :: "i"(n) : "memory")
#endif

// One 16x16x32 WMMA step for both N-halves from a staged LDS tile pair.
// Fragments are two contiguous 16B LDS reads each (ISA 16-bit A/B VGPR layouts:
// lane = row, lane>>4 selects the 8-half K group; VGPR0-3 = K 8h..8h+7,
// VGPR4-7 = K 16+8h..16+8h+7).
__device__ __forceinline__ void wmma_step(
    const __bf16 (*As)[TILE_K + 8], const __bf16 (*Bs)[TILE_K + 8],
    int mrow, int h8, int ncol0, int ncol1, v8f& acc0, v8f& acc1)
{
    v8bf alo = *(const v8bf*)&As[mrow][h8];
    v8bf ahi = *(const v8bf*)&As[mrow][16 + h8];
    v16bf afrag = __builtin_shufflevector(alo, ahi,
        0,1,2,3,4,5,6,7,8,9,10,11,12,13,14,15);
    v8bf b0lo = *(const v8bf*)&Bs[ncol0][h8];
    v8bf b0hi = *(const v8bf*)&Bs[ncol0][16 + h8];
    v16bf b0frag = __builtin_shufflevector(b0lo, b0hi,
        0,1,2,3,4,5,6,7,8,9,10,11,12,13,14,15);
    v8bf b1lo = *(const v8bf*)&Bs[ncol1][h8];
    v8bf b1hi = *(const v8bf*)&Bs[ncol1][16 + h8];
    v16bf b1frag = __builtin_shufflevector(b1lo, b1hi,
        0,1,2,3,4,5,6,7,8,9,10,11,12,13,14,15);
    acc0 = __builtin_amdgcn_wmma_f32_16x16x32_bf16(false, afrag, false, b0frag,
                                                   (short)0, acc0, false, false);
    acc1 = __builtin_amdgcn_wmma_f32_16x16x32_bf16(false, afrag, false, b1frag,
                                                   (short)0, acc1, false, false);
}

// Generic batched bf16 GEMM: C[z] = alpha * A[z] x B[z] (+ f32 bias).
// A: row-major [m][k] bf16. B: "Bt" layout [n][k] bf16 (B[k][n] = Bt[n*ldb+k]),
// so both tiles stage into LDS as contiguous 16B async block copies, DOUBLE-
// BUFFERED: tile s+1 streams in (ASYNCcnt) while tile s is consumed by WMMA.
// BIASMODE: 0=none, 1=per-col bias[n], 2=per-row bias[m].
// CMODE: 0 = f32 C[m][n], 1 = bf16 C[m][n], 2 = bf16 TRANSPOSED Ct[n][m]
// (mode 2 exploits the D layout: 8 contiguous M per lane -> one 16B store).
// Assumes M,N % 64 == 0 and K % 64 == 0 (true for all call sites).
template<int CMODE, int BIASMODE>
__global__ __launch_bounds__(256) void gemm_bf16_wmma(
    int K, int Hdim,
    const __bf16* __restrict__ A, int lda, long long sAb, long long sAh,
    const __bf16* __restrict__ Bt, int ldb, long long sBb, long long sBh,
    const float* __restrict__ bias, long long sBiasH,
    float alpha,
    void* __restrict__ Cout, int ldc, long long sCb, long long sCh)
{
    __shared__ __attribute__((aligned(16))) __bf16 As[2][TILE_M][TILE_K + 8];
    __shared__ __attribute__((aligned(16))) __bf16 Bs[2][TILE_N][TILE_K + 8];

    const int z  = blockIdx.z;
    const int bz = z / Hdim;
    const int hz = z % Hdim;
    const __bf16* Ab = A  + bz * sAb + hz * sAh;
    const __bf16* Bb = Bt + bz * sBb + hz * sBh;
    const float* biasb = bias ? (bias + hz * sBiasH) : nullptr;

    const int m0 = blockIdx.x * TILE_M;
    const int n0 = blockIdx.y * TILE_N;

    const int tid  = threadIdx.x;        // 0..255 (8 waves of 32)
    const int wave = tid >> 5;
    const int lane = tid & 31;

    // 4 waves along M (16 rows) x 2 waves along N (32 cols -> 2 accumulators)
    const int wm = (wave >> 1) * 16;
    const int wn = (wave &  1) * 32;

    // Staging: each thread owns one 16B chunk of the A tile and one of the B tile
    const int sr = tid >> 2;             // tile row 0..63
    const int sc = (tid & 3) * 8;        // halves 0,8,16,24
    const __bf16* aSrc = Ab + (long long)(m0 + sr) * lda + sc;
    const __bf16* bSrc = Bb + (long long)(n0 + sr) * ldb + sc;
    __bf16* aDst0 = &As[0][sr][sc];  __bf16* aDst1 = &As[1][sr][sc];
    __bf16* bDst0 = &Bs[0][sr][sc];  __bf16* bDst1 = &Bs[1][sr][sc];

    v8f acc0 = {};
    v8f acc1 = {};

    const int mrow  = wm + (lane & 15);
    const int h8    = (lane >> 4) * 8;   // half-select * 8
    const int ncol0 = wn + (lane & 15);
    const int ncol1 = wn + 16 + (lane & 15);

    // ---- prime buffer 0 with tile k=0 ----
    ASYNC_CP16(aSrc, aDst0);
    ASYNC_CP16(bSrc, bDst0);

    for (int k0 = 0; k0 < K; k0 += 2 * TILE_K) {
        // phase 0: prefetch tile k0+32 into buf1, compute buf0 (wait oldest 2)
        ASYNC_CP16(aSrc + k0 + TILE_K, aDst1);
        ASYNC_CP16(bSrc + k0 + TILE_K, bDst1);
        ASYNC_WAIT(2);
        __syncthreads();
        wmma_step(As[0], Bs[0], mrow, h8, ncol0, ncol1, acc0, acc1);
        __syncthreads();

        // phase 1: prefetch tile k0+64 into buf0 (unless last), compute buf1
        if (k0 + 2 * TILE_K < K) {
            ASYNC_CP16(aSrc + k0 + 2 * TILE_K, aDst0);
            ASYNC_CP16(bSrc + k0 + 2 * TILE_K, bDst0);
            ASYNC_WAIT(2);
        } else {
            ASYNC_WAIT(0);
        }
        __syncthreads();
        wmma_step(As[1], Bs[1], mrow, h8, ncol0, ncol1, acc0, acc1);
        __syncthreads();
    }

    // ---- epilogue: D layout has N = lane%16 fixed, M = mbase..mbase+7 per lane
    const int nlocal = lane & 15;
    const int mbase  = m0 + wm + 8 * (lane >> 4);
    const int gn0    = n0 + wn + nlocal;
    const int gn1    = gn0 + 16;

    if constexpr (CMODE == 2) {
        // bf16 transposed: Ct[n][m], 8 contiguous m -> one 16B store per acc
        __bf16* Ct = (__bf16*)Cout + bz * sCb + hz * sCh;
        v8bf p0, p1;
        #pragma unroll
        for (int i = 0; i < 8; ++i) {
            const int m = mbase + i;
            float v0 = acc0[i] * alpha, v1 = acc1[i] * alpha;
            if constexpr (BIASMODE == 1) { v0 += biasb[gn0]; v1 += biasb[gn1]; }
            if constexpr (BIASMODE == 2) { v0 += biasb[m];   v1 += biasb[m];   }
            p0[i] = (__bf16)v0; p1[i] = (__bf16)v1;
        }
        *(v8bf*)&Ct[(long long)gn0 * ldc + mbase] = p0;
        *(v8bf*)&Ct[(long long)gn1 * ldc + mbase] = p1;
    } else if constexpr (CMODE == 1) {
        __bf16* Cb = (__bf16*)Cout + bz * sCb + hz * sCh;
        #pragma unroll
        for (int i = 0; i < 8; ++i) {
            const int m = mbase + i;
            float v0 = acc0[i] * alpha, v1 = acc1[i] * alpha;
            if constexpr (BIASMODE == 1) { v0 += biasb[gn0]; v1 += biasb[gn1]; }
            if constexpr (BIASMODE == 2) { v0 += biasb[m];   v1 += biasb[m];   }
            Cb[(long long)m * ldc + gn0] = (__bf16)v0;
            Cb[(long long)m * ldc + gn1] = (__bf16)v1;
        }
    } else {
        float* Cb = (float*)Cout + bz * sCb + hz * sCh;
        #pragma unroll
        for (int i = 0; i < 8; ++i) {
            const int m = mbase + i;
            float v0 = acc0[i] * alpha, v1 = acc1[i] * alpha;
            if constexpr (BIASMODE == 1) { v0 += biasb[gn0]; v1 += biasb[gn1]; }
            if constexpr (BIASMODE == 2) { v0 += biasb[m];   v1 += biasb[m];   }
            Cb[(long long)m * ldc + gn0] = v0;
            Cb[(long long)m * ldc + gn1] = v1;
        }
    }
}

// fp32 -> bf16 elementwise (n % 4 == 0)
__global__ __launch_bounds__(256) void cvt_f32_bf16_kernel(
    const float* __restrict__ src, __bf16* __restrict__ dst, long long n4)
{
    long long i = (long long)blockIdx.x * blockDim.x + threadIdx.x;
    if (i >= n4) return;
    v4f v = *(const v4f*)(src + 4 * i);
    v4bf p; p[0] = (__bf16)v[0]; p[1] = (__bf16)v[1];
            p[2] = (__bf16)v[2]; p[3] = (__bf16)v[3];
    *(v4bf*)(dst + 4 * i) = p;
}

// Softmax over rows of 64: f32 in, bf16 out; one wave per row.
__global__ __launch_bounds__(256) void softmax64_kernel(
    const float* __restrict__ src, __bf16* __restrict__ dst, int rows)
{
    const int row  = blockIdx.x * (blockDim.x >> 5) + (threadIdx.x >> 5);
    if (row >= rows) return;
    const int lane = threadIdx.x & 31;
    const float* r = src + (long long)row * 64;
    float a = r[lane], b = r[lane + 32];
    float mx = fmaxf(a, b);
    #pragma unroll
    for (int off = 16; off; off >>= 1) mx = fmaxf(mx, __shfl_xor(mx, off, 32));
    float ea = __expf(a - mx), eb = __expf(b - mx);
    float s = ea + eb;
    #pragma unroll
    for (int off = 16; off; off >>= 1) s += __shfl_xor(s, off, 32);
    const float inv = 1.0f / s;
    __bf16* w = dst + (long long)row * 64;
    w[lane]      = (__bf16)(ea * inv);
    w[lane + 32] = (__bf16)(eb * inv);
}

extern "C" void kernel_launch(void* const* d_in, const int* in_sizes, int n_in,
                              void* d_out, int out_size, void* d_ws, size_t ws_size,
                              hipStream_t stream)
{
    constexpr int B = 2, N = 4096, C = 1024, H = 16, D = 64;
    const float* x   = (const float*)d_in[0];
    const float* Wtq = (const float*)d_in[1];
    const float* Wtk = (const float*)d_in[2];
    const float* Wtv = (const float*)d_in[3];
    const float* Wq  = (const float*)d_in[4];
    const float* bq  = (const float*)d_in[5];
    const float* Wk  = (const float*)d_in[6];
    const float* bk  = (const float*)d_in[7];
    const float* Wv  = (const float*)d_in[8];
    const float* bv  = (const float*)d_in[9];
    const float* We  = (const float*)d_in[10];
    const float* be  = (const float*)d_in[11];
    const float* Wo  = (const float*)d_in[12];
    const float* bo  = (const float*)d_in[13];
    float* out = (float*)d_out;

    // ---- workspace carve (bf16 intermediates; liveness-based aliasing) ----
    size_t off = 0;
    auto carveB = [&](long long elems) -> __bf16* {
        __bf16* p = (__bf16*)((char*)d_ws + off);
        off += ((size_t)elems * 2 + 255) & ~(size_t)255;
        return p;
    };
    const long long BN   = (long long)B * N;            // 8192
    const long long BHND = (long long)B * H * N * D;    // 8,388,608
    __bf16* xb   = carveB(BN * C);          // later reused as attnb
    __bf16* Wtqb = carveB((long long)D * C);
    __bf16* Wtkb = carveB((long long)D * C);
    __bf16* Wtvb = carveB((long long)D * C);
    __bf16* Wqb  = carveB((long long)H * D * D);
    __bf16* Wkb  = carveB((long long)H * D * D);
    __bf16* Wvb  = carveB((long long)H * D * D);
    __bf16* Web  = carveB((long long)D * N);
    __bf16* Wob  = carveB((long long)C * H * D);
    __bf16* Q0b  = carveB(BN * D);
    __bf16* K0b  = carveB(BN * D);
    __bf16* V0b  = carveB(BN * D);
    __bf16* QWb  = carveB(BHND);            // later reused as Pb
    __bf16* KWt  = carveB(BHND);            // [B,H,e,N] (transposed)
    __bf16* VWt  = carveB(BHND);            // [B,H,d,N] (transposed)
    __bf16* EKtb = carveB((long long)B * H * D * D);    // [B,H,k,e]
    __bf16* FVt  = carveB((long long)B * H * D * D);    // [B,H,d,k] (transposed)
    float*  scores = (float*)KWt;           // 33.5MB spans KWt+VWt (both dead)
    __bf16* Pb     = QWb;                   // QW dead after scores
    __bf16* attnb  = xb;                    // x dead after Q0/K0/V0

    auto cvt = [&](const float* s, __bf16* d, long long n) {
        long long n4 = n / 4;
        dim3 grid((unsigned)((n4 + 255) / 256)), blk(256);
        hipLaunchKernelGGL(cvt_f32_bf16_kernel, grid, blk, 0, stream, s, d, n4);
    };
    #define GEMM(CM, BM, M_, N_, K_, Hd, A_, lda_, sAb_, sAh_,                 \
                 B_, ldb_, sBb_, sBh_, bias_, sBiasH_, alpha_,                 \
                 C_, ldc_, sCb_, sCh_, Z_)                                     \
        hipLaunchKernelGGL((gemm_bf16_wmma<CM, BM>),                           \
                           dim3((M_) / TILE_M, (N_) / TILE_N, (Z_)), dim3(256),\
                           0, stream, (K_), (Hd), (A_), (lda_),                \
                           (long long)(sAb_), (long long)(sAh_),               \
                           (B_), (ldb_), (long long)(sBb_), (long long)(sBh_), \
                           (bias_), (long long)(sBiasH_), (alpha_),            \
                           (void*)(C_), (ldc_), (long long)(sCb_),             \
                           (long long)(sCh_))

    // ---- 0) one-time fp32 -> bf16 conversions ----
    cvt(x, xb, BN * C);
    cvt(Wtq, Wtqb, (long long)D * C);  cvt(Wtk, Wtkb, (long long)D * C);
    cvt(Wtv, Wtvb, (long long)D * C);
    cvt(Wq, Wqb, (long long)H * D * D); cvt(Wk, Wkb, (long long)H * D * D);
    cvt(Wv, Wvb, (long long)H * D * D);
    cvt(We, Web, (long long)D * N);
    cvt(Wo, Wob, (long long)C * H * D);

    const long long sNh = (long long)N * D;   // head stride in [B,H,N,D]
    const long long sNb = (long long)H * N * D;
    const long long sEh = (long long)D * D;
    const long long sEb = (long long)H * D * D;

    // 1) Q0/K0/V0 = x @ Wt*^T   [8192x1024]x[1024x64] -> bf16
    GEMM(1, 0, 8192, 64, 1024, 1, xb, C, 0, 0, Wtqb, C, 0, 0, nullptr, 0, 1.0f,
         Q0b, D, 0, 0, 1);
    GEMM(1, 0, 8192, 64, 1024, 1, xb, C, 0, 0, Wtkb, C, 0, 0, nullptr, 0, 1.0f,
         K0b, D, 0, 0, 1);
    GEMM(1, 0, 8192, 64, 1024, 1, xb, C, 0, 0, Wtvb, C, 0, 0, nullptr, 0, 1.0f,
         V0b, D, 0, 0, 1);

    // 2) per-head linears (z = b*H + h). QW row-major; KW/VW written TRANSPOSED
    //    so they are Bt-layout operands for the E/F projections.
    GEMM(1, 1, N, D, D, H, Q0b, D, sNh, 0, Wqb, D, 0, sEh, bq, D, 1.0f,
         QWb, D, sNb, sNh, B * H);
    GEMM(2, 1, N, D, D, H, K0b, D, sNh, 0, Wkb, D, 0, sEh, bk, D, 1.0f,
         KWt, N, sNb, sNh, B * H);                          // KWt[e][n], ldcT=N
    GEMM(2, 1, N, D, D, H, V0b, D, sNh, 0, Wvb, D, 0, sEh, bv, D, 1.0f,
         VWt, N, sNb, sNh, B * H);                          // VWt[d][n]

    // 3) EKt[k,e] = We @ KW + be (row bias); FVt[d,k] = (We @ VW + be)^T  K=4096
    GEMM(1, 2, D, D, N, H, Web, N, 0, 0, KWt, N, sNb, sNh, be, 0, 1.0f,
         EKtb, D, sEb, sEh, B * H);
    GEMM(2, 2, D, D, N, H, Web, N, 0, 0, VWt, N, sNb, sNh, be, 0, 1.0f,
         FVt, D, sEb, sEh, B * H);                          // transposed out

    // 4) scores = (QW @ EKt^T) / 8 -> f32 (overlays KWt+VWt)
    GEMM(0, 0, N, D, D, H, QWb, D, sNb, sNh, EKtb, D, sEb, sEh, nullptr, 0,
         0.125f, scores, D, sNb, sNh, B * H);

    // 5) softmax over k=64: f32 -> bf16 P (overlays QWb)
    {
        const int rows = B * H * N;                         // 131072
        dim3 grid((rows + 7) / 8), blk(256);
        hipLaunchKernelGGL(softmax64_kernel, grid, blk, 0, stream, scores, Pb, rows);
    }

    // 6) head outputs into concat layout [B,N,H*D] bf16 (overlays xb)
    GEMM(1, 0, N, D, D, H, Pb, D, sNb, sNh, FVt, D, sEb, sEh, nullptr, 0, 1.0f,
         attnb, H * D, (long long)N * H * D, (long long)D, B * H);

    // 7) out = attn @ Wo^T + bo -> f32   [8192x1024]x[1024x1024]
    GEMM(0, 1, 8192, C, H * D, 1, attnb, H * D, 0, 0, Wob, H * D, 0, 0, bo, 0,
         1.0f, out, C, 0, 0, 1);
    #undef GEMM
}
